// MyConv2d_57019985821699
// MI455X (gfx1250) — matrix operand order repaired and checked by reading, not compiled, not run on hardware
//
#include <hip/hip_runtime.h>

// ---- CDNA5 WMMA types -------------------------------------------------------
typedef __bf16 v16bf __attribute__((ext_vector_type(16)));
typedef float  v8f   __attribute__((ext_vector_type(8)));

#define C_IN    64
#define K_OUT   64
#define HW      224
#define TILE_W  112         // output pixels per workgroup (7 waves * 16)
#define WAVES   7
#define NTHREAD 224
#define COLS    114         // TILE_W + 2 halo cols
#define CHP     40          // 32 staged channels + pad (80B row stride, 16B aligned)
#define NCHUNK  18          // 9 taps * 2 channel halves (K = 576 = 18 * 32)
#define A_ELEMS (4 * NCHUNK * 32 * 16)   // 36864 bf16 per (hi|lo) array

// ---- Weight prep: fp32 OIHW -> WMMA A-fragment layout, bf16 hi/lo split -----
// fragIdx = ((mt*18 + chunk)*32 + lane)*16 + j
// A 16x32 bf16 layout: lane row M = mt*16 + lane%16,
//   Klocal = j + 16*(lane>=16);  kk = chunk*32 + Klocal
//   chunk = tap*2 + half; cin = half*32 + Klocal; tap -> (r,s)
__global__ void prep_weights(const float* __restrict__ w,
                             __bf16* __restrict__ ahi,
                             __bf16* __restrict__ alo) {
    int idx = blockIdx.x * blockDim.x + threadIdx.x;
    if (idx >= A_ELEMS) return;
    int j    = idx & 15;
    int lane = (idx >> 4) & 31;
    int c    = (idx >> 9) % NCHUNK;
    int mt   = idx / (NCHUNK * 512);

    int kl  = j + 16 * (lane >> 4);
    int t   = c >> 1;
    int cin = (c & 1) * 32 + kl;
    int m   = mt * 16 + (lane & 15);
    int r   = t / 3, s = t - 3 * r;

    float v = w[((m * C_IN + cin) * 3 + r) * 3 + s];
    __bf16 h = (__bf16)v;
    ahi[idx] = h;
    alo[idx] = (__bf16)(v - (float)h);
}

// ---- Main conv: implicit GEMM, bf16x3 split-precision WMMA ------------------
__global__ __launch_bounds__(NTHREAD)
void conv3x3_wmma(const float* __restrict__ x,
                  const __bf16* __restrict__ ahi,
                  const __bf16* __restrict__ alo,
                  float* __restrict__ out) {
    __shared__ __attribute__((aligned(16))) __bf16 lds_hi[3 * COLS * CHP];
    __shared__ __attribute__((aligned(16))) __bf16 lds_lo[3 * COLS * CHP];

    const int n     = blockIdx.z;
    const int h     = blockIdx.y;
    const int wbase = blockIdx.x * TILE_W;
    const int tid   = threadIdx.x;
    const int wave  = tid >> 5;
    const int lane  = tid & 31;
    const int np    = lane & 15;     // B column / output pixel within wave tile
    const int hiH   = lane >> 4;     // 0: K 0-15, 1: K 16-31 (B striping)

    v8f acc[4];
    acc[0] = v8f{}; acc[1] = v8f{}; acc[2] = v8f{}; acc[3] = v8f{};

    for (int hf = 0; hf < 2; ++hf) {            // channel halves (32 each)
        if (hf) __syncthreads();                // drain reads before restage
        // ---- stage 32 channels x 3 rows x 114 cols, channel-minor, hi/lo ----
        for (int i = tid; i < 3 * 32 * COLS; i += NTHREAD) {
            int col = i % COLS;
            int rc  = i / COLS;
            int ch  = rc & 31;
            int r   = rc >> 5;
            int hh  = h + r - 1;
            int ww  = wbase + col - 1;
            float v = 0.0f;
            if (hh >= 0 && hh < HW && ww >= 0 && ww < HW)
                v = x[((n * C_IN + (hf * 32 + ch)) * HW + hh) * HW + ww];
            __bf16 hv = (__bf16)v;
            int o = (r * COLS + col) * CHP + ch;
            lds_hi[o] = hv;
            lds_lo[o] = (__bf16)(v - (float)hv);
        }
        __syncthreads();

        // ---- 9 taps, K=32 chunk per tap-half ----
        #pragma unroll
        for (int t = 0; t < 9; ++t) {
            const int r = t / 3, s = t - 3 * r;
            const int c = t * 2 + hf;                  // global chunk id
            const int colb = wave * 16 + np + s;       // 0..113 (halo handled)
            const int boff = (r * COLS + colb) * CHP + hiH * 16;

            v16bf bh = *(const v16bf*)&lds_hi[boff];   // 2x ds_load_b128
            v16bf bl = *(const v16bf*)&lds_lo[boff];

            #pragma unroll
            for (int mt = 0; mt < 4; ++mt) {
                const int aoff = ((mt * NCHUNK + c) * 32 + lane) * 16;
                v16bf ah = *(const v16bf*)(ahi + aoff);
                v16bf al = *(const v16bf*)(alo + aoff);
                // D += Ahi*Bhi + Ahi*Blo + Alo*Bhi  (fp32 accumulate)
                acc[mt] = __builtin_amdgcn_wmma_f32_16x16x32_bf16(
                    false, ah, false, bh, (short)0, acc[mt], false, false);
                acc[mt] = __builtin_amdgcn_wmma_f32_16x16x32_bf16(
                    false, ah, false, bl, (short)0, acc[mt], false, false);
                acc[mt] = __builtin_amdgcn_wmma_f32_16x16x32_bf16(
                    false, al, false, bh, (short)0, acc[mt], false, false);
            }
        }
    }

    // ---- writeback: C/D layout: VGPR p -> M = 8*hiH + p, N = np ----
    const int wcol = wbase + wave * 16 + np;
    #pragma unroll
    for (int mt = 0; mt < 4; ++mt) {
        #pragma unroll
        for (int p = 0; p < 8; ++p) {
            int ko = mt * 16 + hiH * 8 + p;
            out[((n * K_OUT + ko) * HW + h) * HW + wcol] = acc[mt][p];
        }
    }
}

// ---- launch -----------------------------------------------------------------
extern "C" void kernel_launch(void* const* d_in, const int* in_sizes, int n_in,
                              void* d_out, int out_size, void* d_ws, size_t ws_size,
                              hipStream_t stream) {
    const float* x = (const float*)d_in[0];
    const float* w = (const float*)d_in[1];
    float* out = (float*)d_out;

    __bf16* ahi = (__bf16*)d_ws;            // 36864 bf16
    __bf16* alo = ahi + A_ELEMS;            // 36864 bf16  (total 144 KB of ws)

    prep_weights<<<dim3((A_ELEMS + 255) / 256), dim3(256), 0, stream>>>(w, ahi, alo);
    conv3x3_wmma<<<dim3(2, HW, 16), dim3(NTHREAD), 0, stream>>>(x, ahi, alo, out);
}
